// ANDCriterion_13589276525197
// MI455X (gfx1250) — matrix operand order repaired and checked by reading, not compiled, not run on hardware
//
#include <hip/hip_runtime.h>

typedef __attribute__((ext_vector_type(16))) _Float16 v16h;
typedef __attribute__((ext_vector_type(8)))  _Float16 v8h;
typedef __attribute__((ext_vector_type(8)))  float    v8f;

#define N_ROWS 8192
#define DIMS   1024
#define INV_T  10.0f    // 1 / temperature(0.1)
#define LOGIT_MAX 10.0f // sim <= 1  =>  logits <= 10 : fixed softmax shift
#define NEG_BIG (-1e30f)

// ---------------------------------------------------------------------------
// Kernel 1: row L2-normalize, emit f16 zn into workspace.
// ---------------------------------------------------------------------------
__global__ __launch_bounds__(256) void normalize_f16_kernel(const float* __restrict__ z,
                                                            _Float16* __restrict__ zn) {
  const int row = blockIdx.x;
  const float* zr = z + (size_t)row * DIMS;
  float acc = 0.f;
  for (int k = threadIdx.x; k < DIMS; k += 256) { float v = zr[k]; acc += v * v; }
  __shared__ float red[256];
  red[threadIdx.x] = acc;
  __syncthreads();
  for (int s = 128; s > 0; s >>= 1) {
    if (threadIdx.x < s) red[threadIdx.x] += red[threadIdx.x + s];
    __syncthreads();
  }
  const float scale = 1.0f / fmaxf(sqrtf(red[0]), 1e-12f);
  _Float16* o = zn + (size_t)row * DIMS;
  for (int k = threadIdx.x; k < DIMS; k += 256) o[k] = (_Float16)(zr[k] * scale);
}

// ---------------------------------------------------------------------------
// Kernel 2: fused sim = zn*zn^T (WMMA f16->f32) + fixed-shift softmax sum +
// top-5, single pass; sim never touches global memory.
// One block = 16 rows; 8 waves stream disjoint 16-col tiles over all columns.
// Softmax uses fixed shift LOGIT_MAX (valid since sim<=1): no online rescale,
// so the epilogue is a branch-light VALU stream that co-executes with WMMA.
// ---------------------------------------------------------------------------
__global__ __launch_bounds__(256) void sim_topk_lse_kernel(const _Float16* __restrict__ zn,
                                                           float* __restrict__ row_loss) {
  __shared__ __align__(32) _Float16 ldsA[16 * DIMS];  // 32 KB: A block (16 rows x 1024)
  __shared__ float ldsC[8][16][16];                   // 8 KB: per-wave C tile spill
  __shared__ float ldsSt[16][16][8];                  // 8 KB: 16 partial states x 16 rows

  const int i0 = blockIdx.x * 16;

  // Stage A block into LDS (2048 x uint4 = 32 KB).
  {
    const uint4* src = (const uint4*)(zn + (size_t)i0 * DIMS);
    uint4* dst = (uint4*)ldsA;
    #pragma unroll
    for (int t = 0; t < 8; ++t) dst[threadIdx.x + 256 * t] = src[threadIdx.x + 256 * t];
  }
  __syncthreads();

  const int wave = threadIdx.x >> 5;
  const int lane = threadIdx.x & 31;
  const int hf   = lane >> 4;     // half-wave: selects K/M/N sub-ranges per ISA layout
  const int ln   = lane & 15;
  const int gi   = i0 + ln;       // global row this thread scans
  const int cbase = hf * 8;       // 8 columns of the C tile this thread scans

  // Fixed-shift softmax accumulator + sorted-descending top-5 (sim domain).
  float S = 0.f;
  float t5[5] = {NEG_BIG, NEG_BIG, NEG_BIG, NEG_BIG, NEG_BIG};

  for (int jt = 0; jt < 64; ++jt) {
    const int j0 = (jt * 8 + wave) * 16;

    v8f c = {0.f, 0.f, 0.f, 0.f, 0.f, 0.f, 0.f, 0.f};
    // B fragment (32x16, 16-bit): lane ln = column n -> row j0+n of zn;
    // K = kb + 16*hf + (0..15): 32 contiguous bytes per kb step.
    const _Float16* brow = zn + (size_t)(j0 + ln) * DIMS + hf * 16;
    // A fragment (16x32, 16-bit): lane ln = row m; v0..3: K = kb+16*hf+(0..7),
    // v4..7: K = kb+16*hf+8+(0..7)  -> two 16-byte LDS reads.
    const _Float16* arow = ldsA + ln * DIMS + hf * 16;

    #pragma unroll 4
    for (int kb = 0; kb < DIMS; kb += 32) {
      v16h b = *(const v16h*)(brow + kb);
      v8h alo = *(const v8h*)(arow + kb);
      v8h ahi = *(const v8h*)(arow + kb + 8);
      v16h a;
      #pragma unroll
      for (int q = 0; q < 8; ++q) { a[q] = alo[q]; a[q + 8] = ahi[q]; }
      c = __builtin_amdgcn_wmma_f32_16x16x32_f16(false, a, false, b,
                                                 (short)0, c, false, false);
    }

    // Spill C tile (lane holds rows cbase..cbase+7 at column ln) to LDS, then
    // scan transposed: each lane owns row ln, columns cbase..cbase+7.
    #pragma unroll
    for (int v = 0; v < 8; ++v) ldsC[wave][cbase + v][ln] = c[v];
    asm volatile("s_wait_dscnt 0" ::: "memory");  // intra-wave LDS RAW fence

    #pragma unroll
    for (int q = 0; q < 8; ++q) {
      const int jj = j0 + cbase + q;
      const float s = ldsC[wave][ln][cbase + q];
      const bool offdiag = (jj != gi);            // diagonal masked in reference
      const float e = __expf(__builtin_fmaf(s, INV_T, -LOGIT_MAX));
      S += offdiag ? e : 0.f;                     // cndmask, no divergence
      if (offdiag && s > t5[4]) {                 // sorted insert into top-5
        t5[4] = s;
        #pragma unroll
        for (int k = 4; k > 0; --k)
          if (t5[k] > t5[k - 1]) { float tm = t5[k - 1]; t5[k - 1] = t5[k]; t5[k] = tm; }
      }
    }
  }

  // Publish this thread's partial state (16 partials per row: 8 waves x 2 halves).
  {
    float* st = ldsSt[wave * 2 + hf][ln];
    st[0] = S;
    #pragma unroll
    for (int k = 0; k < 5; ++k) st[1 + k] = t5[k];
  }
  __syncthreads();

  // Deterministic per-row merge + loss (all partials share the fixed shift).
  if (threadIdx.x < 16) {
    const int r = threadIdx.x;
    float Sg = 0.f;
    float g5[5] = {NEG_BIG, NEG_BIG, NEG_BIG, NEG_BIG, NEG_BIG};
    for (int p = 0; p < 16; ++p) {
      Sg += ldsSt[p][r][0];
      #pragma unroll
      for (int k = 0; k < 5; ++k) {
        const float cand = ldsSt[p][r][1 + k];
        if (cand > g5[4]) {
          g5[4] = cand;
          #pragma unroll
          for (int q = 4; q > 0; --q)
            if (g5[q] > g5[q - 1]) { float tm = g5[q - 1]; g5[q - 1] = g5[q]; g5[q] = tm; }
        }
      }
    }
    float S5 = 0.f;
    #pragma unroll
    for (int k = 0; k < 5; ++k)
      S5 += __expf(__builtin_fmaf(g5[k], INV_T, -LOGIT_MAX));
    // -logsumexp(neigh_logp) = log(S_all) - log(S_top5)  (common shift cancels)
    row_loss[i0 + r] = __logf(Sg) - __logf(S5);
  }
}

// ---------------------------------------------------------------------------
// Kernel 3: deterministic mean over 8192 row losses -> scalar.
// ---------------------------------------------------------------------------
__global__ __launch_bounds__(256) void reduce_mean_kernel(const float* __restrict__ row_loss,
                                                          float* __restrict__ out) {
  float acc = 0.f;
  for (int k = threadIdx.x; k < N_ROWS; k += 256) acc += row_loss[k];
  __shared__ float red[256];
  red[threadIdx.x] = acc;
  __syncthreads();
  for (int s = 128; s > 0; s >>= 1) {
    if (threadIdx.x < s) red[threadIdx.x] += red[threadIdx.x + s];
    __syncthreads();
  }
  if (threadIdx.x == 0) out[0] = red[0] * (1.0f / N_ROWS);
}

// ---------------------------------------------------------------------------
extern "C" void kernel_launch(void* const* d_in, const int* in_sizes, int n_in,
                              void* d_out, int out_size, void* d_ws, size_t ws_size,
                              hipStream_t stream) {
  (void)in_sizes; (void)n_in; (void)out_size; (void)ws_size;
  const float* z = (const float*)d_in[0];
  _Float16* zn  = (_Float16*)d_ws;                                   // 16 MB
  float* row_loss = (float*)((char*)d_ws + (size_t)N_ROWS * DIMS * sizeof(_Float16));
  float* out = (float*)d_out;

  normalize_f16_kernel<<<N_ROWS, 256, 0, stream>>>(z, zn);
  sim_topk_lse_kernel<<<N_ROWS / 16, 256, 0, stream>>>(zn, row_loss);
  reduce_mean_kernel<<<1, 256, 0, stream>>>(row_loss, out);
}
